// PointNet_4234837753917
// MI455X (gfx1250) — matrix operand order repaired and checked by reading, not compiled, not run on hardware
//
#include <hip/hip_runtime.h>
#include <hip/hip_bf16.h>
#include <math.h>

// ---------------------------------------------------------------------------
// PointNet encoder for MI455X (gfx1250, wave32, WMMA).
// B=8 graphs, N=1024 pts, K=16 kNN, P=8192 points, R=P*K=131072 MLP rows.
// Dominant GEMM: [131072,128] x [128,1024] (+max over 16-row groups) -> WMMA
// via v_wmma_f32_16x16x32_f16 (f16 A/B, f32 accumulate), with weights
// pre-packed into per-lane fragment layout so B loads are 2x b128 per tile.
// ---------------------------------------------------------------------------

typedef _Float16 half_t;
typedef __attribute__((ext_vector_type(16))) _Float16 v16h;
typedef __attribute__((ext_vector_type(8)))  float    v8f;
typedef __attribute__((ext_vector_type(4)))  float    v4f;

#define NPTS   1024
#define NGRAPH 8
#define KNN    16
#define NPOINT (NGRAPH * NPTS)       // 8192
#define NROWS  (NPOINT * KNN)        // 131072
#define BN_EPS 1e-5f

// ---------------------------------------------------------------------------
// 1) kNN per graph. score(m) = |pm|^2 - 2*<pn,pm>  (same ordering as ref dist)
// ---------------------------------------------------------------------------
__global__ __launch_bounds__(256) void knn_kernel(const float* __restrict__ pos,
                                                  int* __restrict__ idx)
{
    __shared__ float sp[NPTS * 3];
    __shared__ float sq[NPTS];
    const int tid = threadIdx.x;
    const int gp  = blockIdx.x * 256 + tid;   // global point id
    const int b   = gp >> 10;
    const int base = b << 10;
    for (int i = tid; i < NPTS; i += 256) {
        float x = pos[3 * (base + i) + 0];
        float y = pos[3 * (base + i) + 1];
        float z = pos[3 * (base + i) + 2];
        sp[3 * i + 0] = x; sp[3 * i + 1] = y; sp[3 * i + 2] = z;
        sq[i] = x * x + y * y + z * z;
    }
    __syncthreads();

    const int n = gp & (NPTS - 1);
    const float px = sp[3 * n], py = sp[3 * n + 1], pz = sp[3 * n + 2];

    float bd[KNN]; int bi[KNN];
#pragma unroll
    for (int t = 0; t < KNN; ++t) { bd[t] = 1e30f; bi[t] = 0; }

    for (int m = 0; m < NPTS; ++m) {
        float s = sq[m] - 2.0f * (px * sp[3 * m] + py * sp[3 * m + 1] + pz * sp[3 * m + 2]);
        if (s < bd[KNN - 1]) {               // strict < keeps earlier index on ties
            bd[KNN - 1] = s; bi[KNN - 1] = m;
#pragma unroll
            for (int t = KNN - 1; t > 0; --t) {
                if (bd[t] < bd[t - 1]) {
                    float td = bd[t]; bd[t] = bd[t - 1]; bd[t - 1] = td;
                    int   ti = bi[t]; bi[t] = bi[t - 1]; bi[t - 1] = ti;
                }
            }
        }
    }
#pragma unroll
    for (int t = 0; t < KNN; ++t) idx[gp * KNN + t] = base + bi[t];
}

// ---------------------------------------------------------------------------
// 2) conv1 layer1 pre-activation: t1[r][0..63] = [pj,rel] @ W1 + b1 (6->64),
//    also stores rel[r][3]. One thread per row r.
// ---------------------------------------------------------------------------
__global__ __launch_bounds__(256) void conv1_layer1(const float* __restrict__ pos,
                                                    const int* __restrict__ idx,
                                                    const float* __restrict__ W1,
                                                    const float* __restrict__ b1,
                                                    float* __restrict__ rel,
                                                    float* __restrict__ t1)
{
    __shared__ float sW[6 * 64];
    __shared__ float sb[64];
    const int tid = threadIdx.x;
    for (int i = tid; i < 6 * 64; i += 256) sW[i] = W1[i];
    if (tid < 64) sb[tid] = b1[tid];
    __syncthreads();

    const int r = blockIdx.x * 256 + tid;
    const int p = r >> 4;
    const int j = idx[r];
    float pj0 = pos[3 * j], pj1 = pos[3 * j + 1], pj2 = pos[3 * j + 2];
    float pi0 = pos[3 * p], pi1 = pos[3 * p + 1], pi2 = pos[3 * p + 2];
    float f[6] = { pj0, pj1, pj2, pj0 - pi0, pj1 - pi1, pj2 - pi2 };
    rel[3 * r + 0] = f[3]; rel[3 * r + 1] = f[4]; rel[3 * r + 2] = f[5];

    float* out = t1 + (size_t)r * 64;
    for (int c = 0; c < 64; ++c) {
        float a = sb[c];
#pragma unroll
        for (int ff = 0; ff < 6; ++ff) a += f[ff] * sW[ff * 64 + c];
        out[c] = a;
    }
}

// ---------------------------------------------------------------------------
// 3) BN statistics (deterministic tree reduce). One block per channel.
//    Emits fused affine: bn(x) = scale[c]*x + shift[c].
// ---------------------------------------------------------------------------
__global__ __launch_bounds__(256) void bn_stats(const float* __restrict__ src,
                                                int rows, int C,
                                                const float* __restrict__ gamma,
                                                const float* __restrict__ beta,
                                                float* __restrict__ scale,
                                                float* __restrict__ shift)
{
    __shared__ float s1[256], s2[256];
    const int c = blockIdx.x;
    float sum = 0.f, sq = 0.f;
    for (int r = threadIdx.x; r < rows; r += 256) {
        float v = src[(size_t)r * C + c];
        sum += v; sq += v * v;
    }
    s1[threadIdx.x] = sum; s2[threadIdx.x] = sq;
    __syncthreads();
    for (int off = 128; off > 0; off >>= 1) {
        if (threadIdx.x < off) {
            s1[threadIdx.x] += s1[threadIdx.x + off];
            s2[threadIdx.x] += s2[threadIdx.x + off];
        }
        __syncthreads();
    }
    if (threadIdx.x == 0) {
        float inv  = 1.0f / (float)rows;
        float mean = s1[0] * inv;
        float var  = s2[0] * inv - mean * mean;
        float rstd = rsqrtf(var + BN_EPS);
        scale[c] = gamma[c] * rstd;
        shift[c] = beta[c] - gamma[c] * rstd * mean;
    }
}

// ---------------------------------------------------------------------------
// 4) Pack f32 weights [kdim x ncols] into f16 WMMA B-fragment layout:
//    Wp[((tile*KBLKS + kb)*32 + lane)*16 + e] = W[kb*32 + 16*(lane>>4) + e]
//                                                [tile*16 + (lane&15)]
//    so each lane's v16h fragment is one contiguous 32-byte block.
// ---------------------------------------------------------------------------
__global__ __launch_bounds__(256) void pack_weights(const float* __restrict__ W,
                                                    half_t* __restrict__ Wp,
                                                    int kdim, int ncols)
{
    const int i = blockIdx.x * 256 + threadIdx.x;
    if (i >= kdim * ncols) return;
    const int kblks = kdim >> 5;
    const int e    = i & 15;
    const int lane = (i >> 4) & 31;
    const int j    = i >> 9;
    const int kb   = j % kblks;
    const int tile = j / kblks;
    const int k    = kb * 32 + 16 * (lane >> 4) + e;
    const int col  = tile * 16 + (lane & 15);
    Wp[i] = (half_t)W[(size_t)k * ncols + col];
}

// ---------------------------------------------------------------------------
// 5) WMMA layer: out = [relu(scale*A+shift)] @ W + bias, optional max over
//    each 16-row group (== one point's K=16 neighborhood).
//    One wave handles 2 row-tiles x 4 column-tiles: B fragments and BN
//    vectors are reused across both A tiles (8 wmma per K-step).
//    Fragment layouts used here (A and B share contiguous k order, not the ISA table order):
//      A f16 16x32: lane L (M=L%16, khi=L/16): elem e -> K=16*khi+e,
//                   (one contiguous 16-f32 run, read as 4x v4f)
//      B f16 32x16: lane L (N=L%16): elem e -> K=16*khi+e (pre-packed v16h)
//      D f32 16x16: VGPR i -> M=i+8*khi, N=L%16
// ---------------------------------------------------------------------------
template <int KDIM, bool DO_MAX>
__global__ __launch_bounds__(32) void wmma_layer(const float* __restrict__ Asrc,
                                                 const float* __restrict__ scale,
                                                 const float* __restrict__ shift,
                                                 const half_t* __restrict__ Wp,
                                                 const float* __restrict__ bias,
                                                 float* __restrict__ out,
                                                 int ncols)
{
    constexpr int KBLKS = KDIM / 32;
    const int lane  = threadIdx.x & 31;
    const int M     = lane & 15;
    const int khi   = lane >> 4;                 // 0 or 1 (wave half)
    const int rt0   = blockIdx.x * 2;            // two 16-row tiles (two points)
    const int tile0 = blockIdx.y * 4;            // four 16-col tiles

    const v16h* WpV = (const v16h*)Wp;
    const v4f*  sc4 = (const v4f*)scale;
    const v4f*  sh4 = (const v4f*)shift;

    v8f acc[2][4] = {};
    const float* arow0 = Asrc + (size_t)(rt0 * 16 + M) * KDIM;
    const float* arow1 = arow0 + (size_t)16 * KDIM;

#pragma unroll
    for (int kb = 0; kb < KBLKS; ++kb) {
        const int i0 = (kb * 32 + 16 * khi) >> 2;       // v4f index, elems 0..7
        const int i1 = (kb * 32 + 16 * khi + 8) >> 2;   // v4f index, elems 8..15
        v4f s0 = sc4[i0], s1 = sc4[i0 + 1], s2 = sc4[i1], s3 = sc4[i1 + 1];
        v4f h0 = sh4[i0], h1 = sh4[i0 + 1], h2 = sh4[i1], h3 = sh4[i1 + 1];

        v16h afrag[2];
#pragma unroll
        for (int rr = 0; rr < 2; ++rr) {
            const v4f* ar = (const v4f*)(rr ? arow1 : arow0);
            v4f a0 = ar[i0], a1 = ar[i0 + 1], a2 = ar[i1], a3 = ar[i1 + 1];
#pragma unroll
            for (int e = 0; e < 4; ++e) {
                float v;
                v = a0[e] * s0[e] + h0[e]; afrag[rr][e]      = (half_t)(v > 0.f ? v : 0.f);
                v = a1[e] * s1[e] + h1[e]; afrag[rr][e + 4]  = (half_t)(v > 0.f ? v : 0.f);
                v = a2[e] * s2[e] + h2[e]; afrag[rr][e + 8]  = (half_t)(v > 0.f ? v : 0.f);
                v = a3[e] * s3[e] + h3[e]; afrag[rr][e + 12] = (half_t)(v > 0.f ? v : 0.f);
            }
        }
#pragma unroll
        for (int t = 0; t < 4; ++t) {
            v16h bfrag = WpV[(size_t)((tile0 + t) * KBLKS + kb) * 32 + lane];
            acc[0][t] = __builtin_amdgcn_wmma_f32_16x16x32_f16(
                false, afrag[0], false, bfrag, (short)0, acc[0][t], false, false);
            acc[1][t] = __builtin_amdgcn_wmma_f32_16x16x32_f16(
                false, afrag[1], false, bfrag, (short)0, acc[1][t], false, false);
        }
    }

    if (DO_MAX) {
        // max over the 16 rows of each tile: 8 regs + cross-half shuffle.
#pragma unroll
        for (int rr = 0; rr < 2; ++rr) {
#pragma unroll
            for (int t = 0; t < 4; ++t) {
                float m = acc[rr][t][0];
#pragma unroll
                for (int i = 1; i < 8; ++i) m = fmaxf(m, acc[rr][t][i]);
                m = fmaxf(m, __shfl_xor(m, 16, 32));
                if (lane < 16) {
                    int col = (tile0 + t) * 16 + lane;
                    out[(size_t)(rt0 + rr) * ncols + col] = m + bias[col];
                }
            }
        }
    } else {
#pragma unroll
        for (int rr = 0; rr < 2; ++rr) {
#pragma unroll
            for (int t = 0; t < 4; ++t) {
                int col = (tile0 + t) * 16 + M;
#pragma unroll
                for (int i = 0; i < 8; ++i) {
                    int r = (rt0 + rr) * 16 + i + 8 * khi;
                    out[(size_t)r * ncols + col] = acc[rr][t][i] + bias[col];
                }
            }
        }
    }
}

// ---------------------------------------------------------------------------
// 6) conv2 layer1: t2a[r] = [x1[idx[r]](64), rel[r](3)] @ W(67x64) + b
// ---------------------------------------------------------------------------
__global__ __launch_bounds__(256) void conv2_layer1(const float* __restrict__ x1,
                                                    const int* __restrict__ idx,
                                                    const float* __restrict__ rel,
                                                    const float* __restrict__ W,
                                                    const float* __restrict__ bvec,
                                                    float* __restrict__ t2a)
{
    __shared__ float sW[67 * 64];
    __shared__ float sb[64];
    const int tid = threadIdx.x;
    for (int i = tid; i < 67 * 64; i += 256) sW[i] = W[i];
    if (tid < 64) sb[tid] = bvec[tid];
    __syncthreads();

    const int r = blockIdx.x * 256 + tid;
    const int j = idx[r];
    float f[67];
    const v4f* xr = (const v4f*)(x1 + (size_t)j * 64);
#pragma unroll
    for (int i = 0; i < 16; ++i) {
        v4f q = xr[i];
        f[4 * i] = q[0]; f[4 * i + 1] = q[1]; f[4 * i + 2] = q[2]; f[4 * i + 3] = q[3];
    }
    f[64] = rel[3 * r + 0]; f[65] = rel[3 * r + 1]; f[66] = rel[3 * r + 2];

    float* out = t2a + (size_t)r * 64;
    for (int c = 0; c < 64; ++c) {
        float a = sb[c];
#pragma unroll
        for (int ff = 0; ff < 67; ++ff) a += f[ff] * sW[ff * 64 + c];
        out[c] = a;
    }
}

// ---------------------------------------------------------------------------
// 7) per-graph global max pool: g[b][c] = max_{p<1024} x2[b*1024+p][c]
// ---------------------------------------------------------------------------
__global__ __launch_bounds__(256) void graph_maxpool(const float* __restrict__ x2,
                                                     float* __restrict__ g)
{
    const int b = blockIdx.x >> 2;
    const int c = (blockIdx.x & 3) * 256 + threadIdx.x;
    float m = -1e30f;
    for (int p = 0; p < NPTS; ++p)
        m = fmaxf(m, x2[((size_t)b * NPTS + p) * 1024 + c]);
    g[b * 1024 + c] = m;
}

// ---------------------------------------------------------------------------
// 8) final linear: y[b][c] = g[b] . lin_W[:,c] + lin_b[c]   (8 x 256)
// ---------------------------------------------------------------------------
__global__ __launch_bounds__(256) void final_linear(const float* __restrict__ g,
                                                    const float* __restrict__ W,
                                                    const float* __restrict__ bias,
                                                    float* __restrict__ y)
{
    __shared__ float sg[1024];
    const int bb = blockIdx.x, c = threadIdx.x;
    for (int i = c; i < 1024; i += 256) sg[i] = g[bb * 1024 + i];
    __syncthreads();
    float a = bias[c];
    for (int f = 0; f < 1024; ++f) a += sg[f] * W[(size_t)f * 256 + c];
    y[bb * 256 + c] = a;
}

// ---------------------------------------------------------------------------
// 9) final BN over B=8 rows + ReLU -> d_out
// ---------------------------------------------------------------------------
__global__ __launch_bounds__(256) void final_bn(const float* __restrict__ y,
                                                const float* __restrict__ gamma,
                                                const float* __restrict__ beta,
                                                float* __restrict__ out)
{
    const int c = threadIdx.x;
    float v[NGRAPH];
    float mean = 0.f;
#pragma unroll
    for (int b = 0; b < NGRAPH; ++b) { v[b] = y[b * 256 + c]; mean += v[b]; }
    mean *= (1.0f / NGRAPH);
    float var = 0.f;
#pragma unroll
    for (int b = 0; b < NGRAPH; ++b) { float d = v[b] - mean; var += d * d; }
    var *= (1.0f / NGRAPH);
    float rstd = rsqrtf(var + BN_EPS);
#pragma unroll
    for (int b = 0; b < NGRAPH; ++b) {
        float o = gamma[c] * (v[b] - mean) * rstd + beta[c];
        out[b * 256 + c] = o > 0.f ? o : 0.f;
    }
}

// ---------------------------------------------------------------------------
// Launch
// ---------------------------------------------------------------------------
extern "C" void kernel_launch(void* const* d_in, const int* in_sizes, int n_in,
                              void* d_out, int out_size, void* d_ws, size_t ws_size,
                              hipStream_t stream)
{
    (void)in_sizes; (void)n_in; (void)out_size; (void)ws_size;

    const float* pos    = (const float*)d_in[0];
    /* d_in[1] = batch (unused; layout is fixed B x N) */
    const float* c1_W1  = (const float*)d_in[2];
    const float* c1_b1  = (const float*)d_in[3];
    const float* c1_g1  = (const float*)d_in[4];
    const float* c1_be1 = (const float*)d_in[5];
    const float* c1_W2  = (const float*)d_in[6];
    const float* c1_b2  = (const float*)d_in[7];
    const float* c2_W1  = (const float*)d_in[8];
    const float* c2_b1  = (const float*)d_in[9];
    const float* c2_g1  = (const float*)d_in[10];
    const float* c2_be1 = (const float*)d_in[11];
    const float* c2_W2  = (const float*)d_in[12];
    const float* c2_b2  = (const float*)d_in[13];
    const float* c2_g2  = (const float*)d_in[14];
    const float* c2_be2 = (const float*)d_in[15];
    const float* c2_W3  = (const float*)d_in[16];
    const float* c2_b3  = (const float*)d_in[17];
    const float* lin_W  = (const float*)d_in[18];
    const float* lin_b  = (const float*)d_in[19];
    const float* lin_g  = (const float*)d_in[20];
    const float* lin_be = (const float*)d_in[21];

    // ---- workspace layout (bump allocator, 256B aligned) ----
    char* ws = (char*)d_ws;
    size_t off = 0;
    auto bump = [&](size_t bytes) {
        size_t r = off;
        off += (bytes + 255) & ~(size_t)255;
        return r;
    };
    int*    idx    = (int*)   (ws + bump(sizeof(int)   * NROWS * 1));
    float*  rel    = (float*) (ws + bump(sizeof(float) * NROWS * 3));
    float*  t1     = (float*) (ws + bump(sizeof(float) * (size_t)NROWS * 64));  // reused as t2a
    float*  t2b    = (float*) (ws + bump(sizeof(float) * (size_t)NROWS * 128));
    float*  x1     = (float*) (ws + bump(sizeof(float) * (size_t)NPOINT * 64));
    float*  x2     = (float*) (ws + bump(sizeof(float) * (size_t)NPOINT * 1024));
    float*  gpool  = (float*) (ws + bump(sizeof(float) * NGRAPH * 1024));
    float*  ybuf   = (float*) (ws + bump(sizeof(float) * NGRAPH * 256));
    float*  stats  = (float*) (ws + bump(sizeof(float) * 512));
    half_t* hW1b   = (half_t*)(ws + bump(sizeof(half_t) * 64 * 64));
    half_t* hW2b   = (half_t*)(ws + bump(sizeof(half_t) * 64 * 128));
    half_t* hW3    = (half_t*)(ws + bump(sizeof(half_t) * 128 * 1024));
    float* scale1  = stats + 0,   *shift1  = stats + 64;
    float* scale2a = stats + 128, *shift2a = stats + 192;
    float* scale2b = stats + 256, *shift2b = stats + 384;
    float* t2a = t1;  // t1 dead after x1 is produced

    // ---- pipeline (stream-ordered; graph-capture safe) ----
    knn_kernel<<<NPOINT / 256, 256, 0, stream>>>(pos, idx);

    conv1_layer1<<<NROWS / 256, 256, 0, stream>>>(pos, idx, c1_W1, c1_b1, rel, t1);
    bn_stats<<<64, 256, 0, stream>>>(t1, NROWS, 64, c1_g1, c1_be1, scale1, shift1);
    pack_weights<<<(64 * 64 + 255) / 256, 256, 0, stream>>>(c1_W2, hW1b, 64, 64);
    wmma_layer<64, true><<<dim3(NROWS / 32, 1), 32, 0, stream>>>(
        t1, scale1, shift1, hW1b, c1_b2, x1, 64);

    conv2_layer1<<<NROWS / 256, 256, 0, stream>>>(x1, idx, rel, c2_W1, c2_b1, t2a);
    bn_stats<<<64, 256, 0, stream>>>(t2a, NROWS, 64, c2_g1, c2_be1, scale2a, shift2a);
    pack_weights<<<(64 * 128 + 255) / 256, 256, 0, stream>>>(c2_W2, hW2b, 64, 128);
    wmma_layer<64, false><<<dim3(NROWS / 32, 2), 32, 0, stream>>>(
        t2a, scale2a, shift2a, hW2b, c2_b2, t2b, 128);

    bn_stats<<<128, 256, 0, stream>>>(t2b, NROWS, 128, c2_g2, c2_be2, scale2b, shift2b);
    pack_weights<<<(128 * 1024 + 255) / 256, 256, 0, stream>>>(c2_W3, hW3, 128, 1024);
    wmma_layer<128, true><<<dim3(NROWS / 32, 16), 32, 0, stream>>>(
        t2b, scale2b, shift2b, hW3, c2_b3, x2, 1024);

    graph_maxpool<<<NGRAPH * 4, 256, 0, stream>>>(x2, gpool);
    final_linear<<<NGRAPH, 256, 0, stream>>>(gpool, lin_W, lin_b, ybuf);
    final_bn<<<1, 256, 0, stream>>>(ybuf, lin_g, lin_be, (float*)d_out);
}
